// Descriptor_5274219839800
// MI455X (gfx1250) — matrix operand order repaired and checked by reading, not compiled, not run on hardware
//
#include <hip/hip_runtime.h>
#include <hip/hip_bf16.h>
#include <math.h>

typedef __attribute__((ext_vector_type(16))) _Float16 v16h;
typedef __attribute__((ext_vector_type(8)))  _Float16 v8h;
typedef __attribute__((ext_vector_type(8)))  float    v8f;

#define MAXK 64      // max neighbors
#define HDIM 128     // hidden width
#define FDIM 64      // input feature width
#define KPAD 96      // 67 (F+3) padded to 3*32 for K-steps
#define R2   0.04f   // 0.2^2

// ------------------------------------------------------------------
// Kernel 0: pack W1 (67x128, K-padded to 96) and W2 (128x128) into
// f16 WMMA B-fragment layout: frag[(kt*8+nt)*32 + lane][i] with
// k = kt*32 + (lane>=16 ? 8 : 0) + (i&7) + (i>=8 ? 16 : 0), n = nt*16 + (lane&15).
// Each lane's 16 halves are contiguous (32B) -> 2x global_load_b128.
// ------------------------------------------------------------------
__global__ void pack_weights_kernel(const float* __restrict__ W1,
                                    const float* __restrict__ W2,
                                    _Float16* __restrict__ w1f,
                                    _Float16* __restrict__ w2f) {
  int t = blockIdx.x * blockDim.x + threadIdx.x;
  if (t < 3 * 8 * 32 * 16) {
    int i    = t & 15;
    int lane = (t >> 4) & 31;
    int nt   = (t >> 9) & 7;
    int kt   = t >> 12;                 // 0..2
    int k = kt * 32 + ((lane >> 4) << 3) + (i & 7) + ((i >> 3) << 4);
    int n = nt * 16 + (lane & 15);
    float v = (k < 67) ? W1[k * HDIM + n] : 0.0f;
    w1f[t] = (_Float16)v;
  }
  int t2 = t - 3 * 8 * 32 * 16;
  if (t2 >= 0 && t2 < 4 * 8 * 32 * 16) {
    int i    = t2 & 15;
    int lane = (t2 >> 4) & 31;
    int nt   = (t2 >> 9) & 7;
    int kt   = t2 >> 12;                // 0..3
    int k = kt * 32 + ((lane >> 4) << 3) + (i & 7) + ((i >> 3) << 4);
    int n = nt * 16 + (lane & 15);
    w2f[t2] = (_Float16)W2[k * HDIM + n];
  }
}

// ------------------------------------------------------------------
// Kernel 1: ball query. One wave32 per query; scans points in index
// order, ballot + prefix-popcount appends the first <=64 valid ids.
// pos/batch (~1.6MB) stay L2-resident -> VALU-bound, early exit.
// ------------------------------------------------------------------
__global__ void ball_query_kernel(const float* __restrict__ pos,
                                  const int* __restrict__ batch,
                                  const int* __restrict__ key_idx,
                                  int N, int M,
                                  int* __restrict__ nbr,
                                  int* __restrict__ cnt) {
  int gtid = blockIdx.x * blockDim.x + threadIdx.x;
  int wave = gtid >> 5;
  int lane = gtid & 31;
  if (wave >= M) return;

  int q = key_idx[wave];
  float qx = pos[q * 3 + 0];
  float qy = pos[q * 3 + 1];
  float qz = pos[q * 3 + 2];
  int   qb = batch[q];

  unsigned long long lmask = (1ull << lane) - 1ull;
  int count = 0;
  for (int j0 = 0; j0 < N && count < MAXK; j0 += 32) {
    int j = j0 + lane;
    bool ok = false;
    if (j < N) {
      float dx = pos[j * 3 + 0] - qx;
      float dy = pos[j * 3 + 1] - qy;
      float dz = pos[j * 3 + 2] - qz;
      ok = (dx * dx + dy * dy + dz * dz <= R2) && (batch[j] == qb);
    }
    unsigned long long m = __ballot(ok);          // wave32: low 32 bits
    int prefix = __popcll(m & lmask);
    if (ok && (count + prefix) < MAXK)
      nbr[(size_t)wave * MAXK + count + prefix] = j;
    count += __popcll(m);
  }
  if (count > MAXK) count = MAXK;
  if (lane == 0) cnt[wave] = count;
}

// ------------------------------------------------------------------
// Kernel 2: per-query MLP via v_wmma_f32_16x16x32_f16 + masked max
// aggregate + L2 normalize. 4 waves; wave w owns neighbor rows
// [16w,16w+16). A-frags from LDS (16B aligned chunks per ISA layout),
// B-frags from pre-packed global weights (coalesced b128 loads).
// ------------------------------------------------------------------
__launch_bounds__(128)
__global__ void mlp_max_kernel(const float* __restrict__ x,
                               const float* __restrict__ pos,
                               const float* __restrict__ bias1,
                               const float* __restrict__ bias2,
                               const int* __restrict__ key_idx,
                               const int* __restrict__ nbr,
                               const int* __restrict__ cntArr,
                               const _Float16* __restrict__ w1f,
                               const _Float16* __restrict__ w2f,
                               float* __restrict__ out) {
  __shared__ __attribute__((aligned(16))) _Float16 feat[MAXK * KPAD]; // 12 KB
  __shared__ __attribute__((aligned(16))) _Float16 h1[MAXK * HDIM];   // 16 KB
  __shared__ float partial[4 * HDIM];
  __shared__ float wsum[4];

  const int m    = blockIdx.x;
  const int tid  = threadIdx.x;
  const int lane = tid & 31;
  const int wave = tid >> 5;
  const int count = cntArr[m];
  const int q = key_idx[m];
  const float qx = pos[q * 3 + 0];
  const float qy = pos[q * 3 + 1];
  const float qz = pos[q * 3 + 2];

  // ---- build features: feat[row][0:64]=x_j, [64:67]=pos_j-pos_i, rest 0
  {
    int row = tid >> 1;        // 0..63 (2 threads per row)
    int hlf = tid & 1;
    int j = (row < count) ? nbr[(size_t)m * MAXK + row] : -1;
    float rx = 0.f, ry = 0.f, rz = 0.f;
    if (j >= 0) {
      rx = pos[j * 3 + 0] - qx;
      ry = pos[j * 3 + 1] - qy;
      rz = pos[j * 3 + 2] - qz;
    }
    int c0 = hlf * 48;
    for (int c = c0; c < c0 + 48; ++c) {
      float v = 0.0f;
      if (j >= 0) {
        if (c < FDIM)      v = x[(long long)j * FDIM + c];
        else if (c == 64)  v = rx;
        else if (c == 65)  v = ry;
        else if (c == 66)  v = rz;
      }
      feat[row * KPAD + c] = (_Float16)v;
    }
  }
  __syncthreads();

  const int mt    = wave;                  // M-tile this wave owns
  const int colD  = lane & 15;             // D-tile column for this lane
  const int hi8   = (lane >> 4) << 3;      // A-layout K phase (0 or 8)
  const int hiRow = (lane >> 4) << 3;      // D-layout row phase (0 or 8)
  const int rowA  = mt * 16 + (lane & 15); // A-fragment row for this lane

  // ---- GEMM1: h1 = relu(feat @ W1 + b1), K = 96 (3 steps)
  for (int nt = 0; nt < 8; ++nt) {
    v8f acc = {};
#pragma unroll
    for (int kt = 0; kt < 3; ++kt) {
      v8h a0 = *(const v8h*)&feat[rowA * KPAD + kt * 32 + hi8];
      v8h a1 = *(const v8h*)&feat[rowA * KPAD + kt * 32 + 16 + hi8];
      const v8h* pb = (const v8h*)&w1f[((kt * 8 + nt) * 32 + lane) * 16];
      v8h b0 = pb[0], b1v = pb[1];
      v16h a, b;
#pragma unroll
      for (int i = 0; i < 8; ++i) {
        a[i] = a0[i]; a[i + 8] = a1[i];
        b[i] = b0[i]; b[i + 8] = b1v[i];
      }
      acc = __builtin_amdgcn_wmma_f32_16x16x32_f16(
          false, a, false, b, (short)0, acc, false, false);
    }
    float bb = bias1[nt * 16 + colD];
#pragma unroll
    for (int r = 0; r < 8; ++r) {
      float v = acc[r] + bb;
      v = v > 0.0f ? v : 0.0f;
      int row = mt * 16 + r + hiRow;
      h1[row * HDIM + nt * 16 + colD] = (_Float16)v;
    }
  }
  __syncthreads();   // (same-wave dependence only, barrier for safety)

  // ---- GEMM2: h2 = h1 @ W2 + b2, K = 128 (4 steps); masked column max
  for (int nt = 0; nt < 8; ++nt) {
    v8f acc = {};
#pragma unroll
    for (int kt = 0; kt < 4; ++kt) {
      v8h a0 = *(const v8h*)&h1[rowA * HDIM + kt * 32 + hi8];
      v8h a1 = *(const v8h*)&h1[rowA * HDIM + kt * 32 + 16 + hi8];
      const v8h* pb = (const v8h*)&w2f[((kt * 8 + nt) * 32 + lane) * 16];
      v8h b0 = pb[0], b1v = pb[1];
      v16h a, b;
#pragma unroll
      for (int i = 0; i < 8; ++i) {
        a[i] = a0[i]; a[i + 8] = a1[i];
        b[i] = b0[i]; b[i + 8] = b1v[i];
      }
      acc = __builtin_amdgcn_wmma_f32_16x16x32_f16(
          false, a, false, b, (short)0, acc, false, false);
    }
    float bb = bias2[nt * 16 + colD];
    float mx = -INFINITY;
#pragma unroll
    for (int r = 0; r < 8; ++r) {
      int row = mt * 16 + r + hiRow;      // neighbor slot index
      float v = acc[r] + bb;
      if (row < count) mx = fmaxf(mx, v);
    }
    // lanes l and l^16 hold the same column, different rows
    mx = fmaxf(mx, __shfl_xor(mx, 16, 32));
    if (lane < 16) partial[wave * HDIM + nt * 16 + colD] = mx;
  }
  __syncthreads();

  // ---- cross-wave max, zero-if-empty, L2 normalize
  float v = fmaxf(fmaxf(partial[0 * HDIM + tid], partial[1 * HDIM + tid]),
                  fmaxf(partial[2 * HDIM + tid], partial[3 * HDIM + tid]));
  if (count == 0) v = 0.0f;
  float v2 = v * v;
#pragma unroll
  for (int off = 16; off > 0; off >>= 1) v2 += __shfl_xor(v2, off, 32);
  if (lane == 0) wsum[wave] = v2;
  __syncthreads();
  float total = wsum[0] + wsum[1] + wsum[2] + wsum[3];
  float nrm = sqrtf(total);
  nrm = nrm > 1e-12f ? nrm : 1e-12f;
  out[(long long)m * HDIM + tid] = v / nrm;
}

// ------------------------------------------------------------------
extern "C" void kernel_launch(void* const* d_in, const int* in_sizes, int n_in,
                              void* d_out, int out_size, void* d_ws, size_t ws_size,
                              hipStream_t stream) {
  const float* x       = (const float*)d_in[0];
  const float* pos     = (const float*)d_in[1];
  const float* W1      = (const float*)d_in[2];
  const float* b1      = (const float*)d_in[3];
  const float* W2      = (const float*)d_in[4];
  const float* b2      = (const float*)d_in[5];
  const int*   batch   = (const int*)d_in[6];
  const int*   key_idx = (const int*)d_in[7];

  const int N = in_sizes[1] / 3;   // 100000
  const int M = in_sizes[7];       // 2048

  // workspace layout
  _Float16* w1f = (_Float16*)d_ws;                 // 3*8*32*16 halves (24 KB)
  _Float16* w2f = w1f + 3 * 8 * 32 * 16;           // 4*8*32*16 halves (32 KB)
  int* nbr = (int*)(w2f + 4 * 8 * 32 * 16);        // M*64 ints
  int* cnt = nbr + (size_t)M * MAXK;               // M ints

  int packElems = 3 * 8 * 32 * 16 + 4 * 8 * 32 * 16;
  pack_weights_kernel<<<(packElems + 255) / 256, 256, 0, stream>>>(W1, W2, w1f, w2f);

  ball_query_kernel<<<(M * 32 + 255) / 256, 256, 0, stream>>>(
      pos, batch, key_idx, N, M, nbr, cnt);

  mlp_max_kernel<<<M, 128, 0, stream>>>(
      x, pos, b1, b2, key_idx, nbr, cnt, w1f, w2f, (float*)d_out);
}
